// FourierNet_12094627905687
// MI455X (gfx1250) — compile-verified, matrix operand order checked by reading, not run on hardware
//
#include <hip/hip_runtime.h>
#include <hip/hip_bf16.h>
#include <math.h>

// ---------------------------------------------------------------------------
// FourierNet forward for MI455X (gfx1250), wave32, WMMA f32<-f16.
//
// Roofline: ~0.21 TFLOP of GEMM vs ~7 MB of resident weights (fits in L2) and
// ~1 MB of streamed I/O -> compute bound on matrix units. All GEMMs use
// v_wmma_f32_16x16x32_f16; activations staged in LDS in the native WMMA
// A-fragment layout (32B contiguous per lane -> ds_load_b128 pairs).
// Fourier stage minimizes TRANS-pipe work: exp tables in LDS, one sincos pair
// per (row,freq) expanded to all 8 groups via angle-addition FMAs.
// ---------------------------------------------------------------------------

typedef __attribute__((ext_vector_type(16))) _Float16 v16h;
typedef __attribute__((ext_vector_type(8)))  float    v8f;

#define NPTS   65536
#define NS     4
#define NFRQ   64
#define NH     256
#define NBLK   2
#define INF    514      // 2 + 8*64
#define KPAD   544      // INF padded to multiple of 32
#define KC_PROJ 17      // 544/32
#define KC_H    8       // 256/32
#define TM      32      // rows per workgroup (2 WMMA row tiles)

#define S_IN 0.04410811f   // 1/sqrt(514)
#define S_H  0.0625f       // 1/sqrt(256)
#define PI_F 3.14159265358979f

__device__ __forceinline__ float gelu_exact(float z) {
  return 0.5f * z * (1.0f + erff(z * 0.70710678118654752f));
}

// Packed fragment index mapping for 16-bit 16x32 A / 32x16 B fragments:
// within a 32-wide K chunk, lane' = base16 + 16*((krem>>3)&1),
// half index j = (krem&7) | ((krem>>4)<<3).

// ---------------------------------------------------------------------------
// Kernel 1: pack f32 weights -> f16 WMMA B fragments in workspace.
// dest layout: [tileId][kc][lane(32)][j(16)] halves, 32B contiguous per lane.
// ---------------------------------------------------------------------------
__global__ __launch_bounds__(256) void pack_weights_k(
    const float* __restrict__ projW, const float* __restrict__ W1,
    const float* __restrict__ W2, _Float16* __restrict__ pProj,
    _Float16* __restrict__ pW1, _Float16* __restrict__ pW2) {
  const long nProj = (long)NS * 16 * KC_PROJ * 32 * 16;
  const long nBlk  = (long)NS * NBLK * 16 * KC_H * 32 * 16;
  long idx = (long)blockIdx.x * blockDim.x + threadIdx.x;

  if (idx < nProj) {
    int  j    = (int)(idx & 15);
    int  lane = (int)((idx >> 4) & 31);
    long rest = idx >> 9;
    int  kc   = (int)(rest % KC_PROJ); rest /= KC_PROJ;
    int  ct   = (int)(rest & 15);
    int  s    = (int)(rest >> 4);
    int krem = j + ((j >> 3) << 3) + ((lane >> 4) << 3);
    int k    = kc * 32 + krem;
    int col  = ct * 16 + (lane & 15);
    float v = (k < INF) ? projW[((long)(s * NH + col)) * INF + k] : 0.0f;
    pProj[idx] = (_Float16)v;
    return;
  }
  idx -= nProj;
  const float* src;
  _Float16* dst;
  if (idx < nBlk) { src = W1; dst = pW1; }
  else {
    idx -= nBlk;
    if (idx >= nBlk) return;
    src = W2; dst = pW2;
  }
  int  j    = (int)(idx & 15);
  int  lane = (int)((idx >> 4) & 31);
  long rest = idx >> 9;
  int  kc   = (int)(rest & 7);  rest >>= 3;
  int  ct   = (int)(rest & 15); rest >>= 4;
  int  sb   = (int)rest;                       // s*NBLK + b
  int krem = j + ((j >> 3) << 3) + ((lane >> 4) << 3);
  int k    = kc * 32 + krem;
  int col  = ct * 16 + (lane & 15);
  dst[idx] = (_Float16)src[((long)(sb * NH + col)) * NH + k];
}

// ---------------------------------------------------------------------------
// Kernel 2: fused fourier features + 3-GEMM residual MLP per (32 rows, scale).
// ---------------------------------------------------------------------------
__global__ __launch_bounds__(256) void fourier_mlp_k(
    const float* __restrict__ xy,     const float* __restrict__ w_x,
    const float* __restrict__ w_y,    const float* __restrict__ proj_b,
    const float* __restrict__ blk_b1, const float* __restrict__ blk_b2,
    const float* __restrict__ head_W, const float* __restrict__ sc_W,
    const _Float16* __restrict__ pProj, const _Float16* __restrict__ pW1,
    const _Float16* __restrict__ pW2, float* __restrict__ part) {
  // fourier A fragments; reused as the t-activation buffer in res-blocks
  __shared__ _Float16 fA[KC_PROJ][2][32][16];   // 34816 B
  __shared__ _Float16 hA[KC_H][2][32][16];      // 16384 B
  __shared__ float    ysum[TM];
  __shared__ float    exs[NFRQ], eys[NFRQ];     // exp(w_x), exp(w_y) tables
  __shared__ float    scw[KPAD];                // sc_W slice for this scale

  const int tid  = threadIdx.x;
  const int lane = tid & 31;
  const int wave = tid >> 5;
  const int s    = blockIdx.y;
  const int row0 = blockIdx.x * TM;

  if (tid < TM) ysum[tid] = 0.0f;
  if (tid < NFRQ)                exs[tid]        = __expf(w_x[s * NFRQ + tid]);
  else if (tid < 2 * NFRQ)       eys[tid - NFRQ] = __expf(w_y[s * NFRQ + tid - NFRQ]);
  for (int k = tid; k < KPAD; k += 256) scw[k] = (k < INF) ? sc_W[s * INF + k] : 0.0f;
  __syncthreads();

  // ---- stage 1: fourier features -> fA (f16, A layout); shortcut dot ----
  {
    const int m  = tid & 31;            // row within tile (0..31)
    const int rt = m >> 4;
    const int mr = m & 15;
    const int n  = row0 + m;
    const float x = xy[2 * n], y = xy[2 * n + 1];
    float scacc = 0.0f;

    // k = 0,1 raw coords + zero padding region, handled by threads 0..31
    if (tid < TM) {
#pragma unroll
      for (int k = 0; k < 2; ++k) {
        float v = (k == 0) ? x : y;
        scacc += v * scw[k];
        const int lp = mr;                         // krem = k < 8 -> hi bit 0
        const int j  = k;
        fA[0][rt][lp][j] = (_Float16)v;
      }
      for (int k = INF; k < KPAD; ++k) {
        const int krem = k & 31;
        fA[KC_PROJ - 1][rt][mr + (((krem >> 3) & 1) << 4)]
          [(krem & 7) | ((krem >> 4) << 3)] = (_Float16)0.0f;
      }
    }

    const float sx = PI_F * x, sy = PI_F * y;
#pragma unroll
    for (int i = 0; i < 8; ++i) {
      const int f = (tid >> 5) + 8 * i;            // 0..63
      const float px = sx * exs[f], py = sy * eys[f];
      float sp, cp, sq, cq;
      __sincosf(px, &sp, &cp);
      __sincosf(py, &sq, &cq);
      float v[8];
      v[0] = sp;                 // sin(px)
      v[1] = cp;                 // cos(px)
      v[2] = sq;                 // sin(py)
      v[3] = cq;                 // cos(py)
      v[4] = sp * cq + cp * sq;  // sin(px+py)
      v[5] = cp * cq - sp * sq;  // cos(px+py)
      v[6] = sp * cq - cp * sq;  // sin(px-py)
      v[7] = cp * cq + sp * sq;  // cos(px-py)
#pragma unroll
      for (int g = 0; g < 8; ++g) {
        const int k = 2 + g * NFRQ + f;
        scacc += v[g] * scw[k];
        const int kc   = k >> 5;
        const int krem = k & 31;
        fA[kc][rt][mr + (((krem >> 3) & 1) << 4)]
          [(krem & 7) | ((krem >> 4) << 3)] = (_Float16)v[g];
      }
    }
    atomicAdd(&ysum[m], scacc);
  }
  __syncthreads();

  // ---- stage 2: proj GEMM [32,544] x [544,256] -> gelu -> hacc ----
  v8f hacc[2][2];   // [col tile t][row tile rt]
  const int hiL = lane >> 4;
#pragma unroll
  for (int t = 0; t < 2; ++t) {
    const int ct = wave * 2 + t;
    v8f c0 = {}, c1 = {};
    const _Float16* wB = pProj + ((long)(s * 16 + ct)) * KC_PROJ * 512 + lane * 16;
#pragma unroll
    for (int kc = 0; kc < KC_PROJ; ++kc) {
      v16h b  = *(const v16h*)(wB + kc * 512);          // B reused by both row tiles
      v16h a0 = *(const v16h*)(&fA[kc][0][lane][0]);
      v16h a1 = *(const v16h*)(&fA[kc][1][lane][0]);
      c0 = __builtin_amdgcn_wmma_f32_16x16x32_f16(false, a0, false, b, (short)0, c0, false, false);
      c1 = __builtin_amdgcn_wmma_f32_16x16x32_f16(false, a1, false, b, (short)0, c1, false, false);
    }
    const int col   = ct * 16 + (lane & 15);
    const float bia = proj_b[s * NH + col];
#pragma unroll
    for (int r = 0; r < 8; ++r) {
      c0[r] = gelu_exact(c0[r] * S_IN + bia);
      c1[r] = gelu_exact(c1[r] * S_IN + bia);
    }
    hacc[t][0] = c0;
    hacc[t][1] = c1;
    // scatter h (f16) into hA in A-fragment layout for the next GEMM
    const int kc2   = col >> 5;
    const int krem  = col & 31;
    const int j     = (krem & 7) | ((krem >> 4) << 3);
    const int lbase = ((krem >> 3) & 1) << 4;
#pragma unroll
    for (int r = 0; r < 8; ++r) {
      hA[kc2][0][lbase + r + 8 * hiL][j] = (_Float16)c0[r];
      hA[kc2][1][lbase + r + 8 * hiL][j] = (_Float16)c1[r];
    }
  }
  __syncthreads();

  // ---- stage 3: residual blocks ----
  for (int blk = 0; blk < NBLK; ++blk) {
    // t = gelu(h @ W1^T * s_h + b1) -> fA (reused as t buffer)
#pragma unroll
    for (int t = 0; t < 2; ++t) {
      const int ct = wave * 2 + t;
      v8f c0 = {}, c1 = {};
      const _Float16* wB = pW1 + ((long)((s * NBLK + blk) * 16 + ct)) * KC_H * 512 + lane * 16;
#pragma unroll
      for (int kc = 0; kc < KC_H; ++kc) {
        v16h b  = *(const v16h*)(wB + kc * 512);
        v16h a0 = *(const v16h*)(&hA[kc][0][lane][0]);
        v16h a1 = *(const v16h*)(&hA[kc][1][lane][0]);
        c0 = __builtin_amdgcn_wmma_f32_16x16x32_f16(false, a0, false, b, (short)0, c0, false, false);
        c1 = __builtin_amdgcn_wmma_f32_16x16x32_f16(false, a1, false, b, (short)0, c1, false, false);
      }
      const int col   = ct * 16 + (lane & 15);
      const float bia = blk_b1[(s * NBLK + blk) * NH + col];
      const int kc2   = col >> 5;
      const int krem  = col & 31;
      const int j     = (krem & 7) | ((krem >> 4) << 3);
      const int lbase = ((krem >> 3) & 1) << 4;
#pragma unroll
      for (int r = 0; r < 8; ++r) {
        float t0 = gelu_exact(c0[r] * S_H + bia);
        float t1 = gelu_exact(c1[r] * S_H + bia);
        fA[kc2][0][lbase + r + 8 * hiL][j] = (_Float16)t0;
        fA[kc2][1][lbase + r + 8 * hiL][j] = (_Float16)t1;
      }
    }
    __syncthreads();   // t buffer complete (and all hA reads done)

    // h += (t @ W2^T) * s_h + b2
#pragma unroll
    for (int t = 0; t < 2; ++t) {
      const int ct = wave * 2 + t;
      v8f c0 = {}, c1 = {};
      const _Float16* wB = pW2 + ((long)((s * NBLK + blk) * 16 + ct)) * KC_H * 512 + lane * 16;
#pragma unroll
      for (int kc = 0; kc < KC_H; ++kc) {
        v16h b  = *(const v16h*)(wB + kc * 512);
        v16h a0 = *(const v16h*)(&fA[kc][0][lane][0]);
        v16h a1 = *(const v16h*)(&fA[kc][1][lane][0]);
        c0 = __builtin_amdgcn_wmma_f32_16x16x32_f16(false, a0, false, b, (short)0, c0, false, false);
        c1 = __builtin_amdgcn_wmma_f32_16x16x32_f16(false, a1, false, b, (short)0, c1, false, false);
      }
      const int col   = ct * 16 + (lane & 15);
      const float bia = blk_b2[(s * NBLK + blk) * NH + col];
#pragma unroll
      for (int r = 0; r < 8; ++r) {
        hacc[t][0][r] = hacc[t][0][r] + c0[r] * S_H + bia;
        hacc[t][1][r] = hacc[t][1][r] + c1[r] * S_H + bia;
      }
    }
    __syncthreads();   // all t-buffer reads done before it is reused/overwritten

    if (blk + 1 < NBLK) {
      // refresh hA from updated hacc for the next block's GEMM
#pragma unroll
      for (int t = 0; t < 2; ++t) {
        const int col   = (wave * 2 + t) * 16 + (lane & 15);
        const int kc2   = col >> 5;
        const int krem  = col & 31;
        const int j     = (krem & 7) | ((krem >> 4) << 3);
        const int lbase = ((krem >> 3) & 1) << 4;
#pragma unroll
        for (int r = 0; r < 8; ++r) {
          hA[kc2][0][lbase + r + 8 * hiL][j] = (_Float16)hacc[t][0][r];
          hA[kc2][1][lbase + r + 8 * hiL][j] = (_Float16)hacc[t][1][r];
        }
      }
      __syncthreads();
    }
  }

  // ---- stage 4: head dot product, combine with shortcut partial ----
#pragma unroll
  for (int t = 0; t < 2; ++t) {
    const int col  = (wave * 2 + t) * 16 + (lane & 15);
    const float hw = head_W[col];
#pragma unroll
    for (int r = 0; r < 8; ++r) {
      atomicAdd(&ysum[r + 8 * hiL],      hacc[t][0][r] * hw);
      atomicAdd(&ysum[16 + r + 8 * hiL], hacc[t][1][r] * hw);
    }
  }
  __syncthreads();
  if (tid < TM) part[(long)s * NPTS + row0 + tid] = ysum[tid];
}

// ---------------------------------------------------------------------------
// Kernel 3: sum per-scale partials + head bias -> d_out
// ---------------------------------------------------------------------------
__global__ __launch_bounds__(256) void reduce_k(const float* __restrict__ part,
                                                const float* __restrict__ head_b,
                                                float* __restrict__ out) {
  int i = blockIdx.x * 256 + threadIdx.x;
  if (i < NPTS) {
    out[i] = head_b[0] + part[i] + part[(long)NPTS + i] +
             part[2L * NPTS + i] + part[3L * NPTS + i];
  }
}

extern "C" void kernel_launch(void* const* d_in, const int* in_sizes, int n_in,
                              void* d_out, int out_size, void* d_ws, size_t ws_size,
                              hipStream_t stream) {
  (void)in_sizes; (void)n_in; (void)out_size; (void)ws_size;
  const float* xy     = (const float*)d_in[0];
  const float* w_x    = (const float*)d_in[1];
  const float* w_y    = (const float*)d_in[2];
  const float* proj_W = (const float*)d_in[3];
  const float* proj_b = (const float*)d_in[4];
  const float* blk_W1 = (const float*)d_in[5];
  const float* blk_b1 = (const float*)d_in[6];
  const float* blk_W2 = (const float*)d_in[7];
  const float* blk_b2 = (const float*)d_in[8];
  const float* head_W = (const float*)d_in[9];
  const float* head_b = (const float*)d_in[10];
  const float* sc_W   = (const float*)d_in[11];

  char* ws = (char*)d_ws;
  _Float16* pProj = (_Float16*)(ws + 0);        // 557056 halves = 1,114,112 B
  _Float16* pW1   = (_Float16*)(ws + 1114112);  // 524288 halves = 1,048,576 B
  _Float16* pW2   = (_Float16*)(ws + 2162688);  // 524288 halves = 1,048,576 B
  float*    part  = (float*)   (ws + 3211264);  // NS*NPTS f32   = 1,048,576 B

  const long nPack = 557056L + 2L * 524288L;
  pack_weights_k<<<(int)((nPack + 255) / 256), 256, 0, stream>>>(
      proj_W, blk_W1, blk_W2, pProj, pW1, pW2);

  dim3 grid(NPTS / TM, NS);
  fourier_mlp_k<<<grid, 256, 0, stream>>>(xy, w_x, w_y, proj_b, blk_b1, blk_b2,
                                          head_W, sc_W, pProj, pW1, pW2, part);

  reduce_k<<<NPTS / 256, 256, 0, stream>>>(part, head_b, (float*)d_out);
}
